// SDFNetwork_28501402976394
// MI455X (gfx1250) — matrix operand order repaired and checked by reading, not compiled, not run on hardware
//
#include <hip/hip_runtime.h>
#include <hip/hip_bf16.h>

typedef __attribute__((ext_vector_type(16))) _Float16 v16h;
typedef __attribute__((ext_vector_type(8)))  _Float16 v8h;
typedef __attribute__((ext_vector_type(8)))  float    v8f;

#define NLV    16
#define LOG2T  19
#define TSZ    (1u << LOG2T)
#define HID    64
#define STRIDE 72   // padded LDS row stride (halves): 144 B = 9*16 B, keeps b128 loads aligned, spreads banks

// floor(16 * (2048/16)^(l/15))
__constant__ float kRes[NLV] = {16.f, 22.f, 30.f, 42.f, 58.f, 80.f, 111.f, 153.f,
                                212.f, 294.f, 406.f, 561.f, 776.f, 1072.f, 1482.f, 2048.f};

__device__ __forceinline__ v16h make16(v8h lo, v8h hi) {
  return __builtin_shufflevector(lo, hi, 0, 1, 2, 3, 4, 5, 6, 7, 8, 9, 10, 11, 12, 13, 14, 15);
}

// B fragment (K=32 chunk): lane l -> n = tile*16 + (l&15), half = l>>4; elems 0..7 = K[half*8..], 8..15 = K[16+half*8..]
__device__ __forceinline__ v16h load_b(const _Float16* W, int kchunk, int tile, int lane) {
  const int h = lane >> 4;
  const _Float16* p = W + (kchunk * 32) * HID + tile * 16 + (lane & 15);
  v16h r;
  #pragma unroll
  for (int s = 0; s < 16; ++s) {
    const int k = (s < 8) ? (h * 8 + s) : (16 + h * 8 + (s - 8));
    r[s] = p[k * HID];
  }
  return r;
}

// A fragment (16x32 f16): lane l -> m = l&15, half = l>>4; elems 0..7 = K[half*8..], 8..15 = K[16+half*8..]
__device__ __forceinline__ v16h load_a(const _Float16* st, int kchunk, int lane) {
  const int m = lane & 15, h = lane >> 4;
  const _Float16* row = st + m * STRIDE + kchunk * 32;
  v8h lo = *(const v8h*)(row + h * 8);
  v8h hi = *(const v8h*)(row + 16 + h * 8);
  return make16(lo, hi);
}

__global__ __launch_bounds__(256) void sdf_fused(
    const float* __restrict__ x, const float* __restrict__ table,
    const float* __restrict__ w1, const float* __restrict__ b1,
    const float* __restrict__ w2, const float* __restrict__ b2,
    const float* __restrict__ w3, const float* __restrict__ b3,
    const float* __restrict__ w4, const float* __restrict__ b4,
    float* __restrict__ out, int N)
{
  __shared__ alignas(16) _Float16 sW1[32 * HID];
  __shared__ alignas(16) _Float16 sW2[HID * HID];
  __shared__ alignas(16) _Float16 sW3[HID * HID];
  __shared__ alignas(16) _Float16 sStage[8][16 * STRIDE];

  const int tid = threadIdx.x;
  for (int i = tid; i < 32 * HID;  i += 256) sW1[i] = (_Float16)w1[i];
  for (int i = tid; i < HID * HID; i += 256) sW2[i] = (_Float16)w2[i];
  for (int i = tid; i < HID * HID; i += 256) sW3[i] = (_Float16)w3[i];
  __syncthreads();

  const int lane  = tid & 31;
  const int wave  = tid >> 5;
  const int nlane = lane & 15;   // point-in-tile (for encode) / N column (for C/D frags)
  const int hgrp  = lane >> 4;

  const int base = (blockIdx.x * 8 + wave) * 16;
  const int p    = base + nlane;
  const int pc   = p < N ? p : N - 1;  // clamp for loads; stores are guarded

  // -------- hash-grid encode --------
  // Lane computes 16 of its point's 32 features: exactly the f16 A-matrix fragment layout.
  const float px = x[3 * pc + 0], py = x[3 * pc + 1], pz = x[3 * pc + 2];

  v16h afeat;
  #pragma unroll
  for (int g = 0; g < 2; ++g) {
    #pragma unroll
    for (int j = 0; j < 4; ++j) {
      const int lev = g * 8 + hgrp * 4 + j;
      const float res = kRes[lev];
      const float fx = px * res, fy = py * res, fz = pz * res;
      const float x0 = floorf(fx), y0 = floorf(fy), z0 = floorf(fz);
      const float rx = fx - x0, ry = fy - y0, rz = fz - z0;
      const unsigned ux = (unsigned)(int)x0, uy = (unsigned)(int)y0, uz = (unsigned)(int)z0;
      const float* tab = table + (size_t)lev * (size_t)(TSZ * 2);
      float f0 = 0.f, f1 = 0.f;
      #pragma unroll
      for (int c = 0; c < 8; ++c) {
        const unsigned cx = (c >> 2) & 1, cy = (c >> 1) & 1, cz = c & 1;
        const unsigned h = (ux + cx) ^ ((uy + cy) * 2654435761u) ^ ((uz + cz) * 805459861u);
        const unsigned idx = h & (TSZ - 1u);
        const float wx = cx ? rx : 1.f - rx;
        const float wy = cy ? ry : 1.f - ry;
        const float wz = cz ? rz : 1.f - rz;
        const float w  = wx * wy * wz;
        const float2 f = *(const float2*)(tab + 2u * idx);  // L2-resident gather
        f0 += w * f.x; f1 += w * f.y;
      }
      afeat[g * 8 + 2 * j + 0] = (_Float16)f0;
      afeat[g * 8 + 2 * j + 1] = (_Float16)f1;
    }
  }

  _Float16* st = sStage[wave];

  // -------- layer 1: (16x32) @ (32x64), 4 WMMAs --------
  #pragma unroll
  for (int t = 0; t < 4; ++t) {
    const float bias = b1[t * 16 + nlane];
    v8f acc = {bias, bias, bias, bias, bias, bias, bias, bias};
    const v16h bf = load_b(sW1, 0, t, lane);
    acc = __builtin_amdgcn_wmma_f32_16x16x32_f16(false, afeat, false, bf, (short)0, acc, false, false);
    #pragma unroll
    for (int r = 0; r < 8; ++r) {
      const float v = acc[r] > 0.f ? acc[r] : 0.f;
      st[(r + 8 * hgrp) * STRIDE + t * 16 + nlane] = (_Float16)v;
    }
  }
  asm volatile("s_wait_dscnt 0" ::: "memory");

  // -------- layer 2: (16x64) @ (64x64), 8 WMMAs --------
  v16h a0 = load_a(st, 0, lane);
  v16h a1 = load_a(st, 1, lane);
  #pragma unroll
  for (int t = 0; t < 4; ++t) {
    const float bias = b2[t * 16 + nlane];
    v8f acc = {bias, bias, bias, bias, bias, bias, bias, bias};
    acc = __builtin_amdgcn_wmma_f32_16x16x32_f16(false, a0, false, load_b(sW2, 0, t, lane), (short)0, acc, false, false);
    acc = __builtin_amdgcn_wmma_f32_16x16x32_f16(false, a1, false, load_b(sW2, 1, t, lane), (short)0, acc, false, false);
    #pragma unroll
    for (int r = 0; r < 8; ++r) {
      const float v = acc[r] > 0.f ? acc[r] : 0.f;
      st[(r + 8 * hgrp) * STRIDE + t * 16 + nlane] = (_Float16)v;
    }
  }
  asm volatile("s_wait_dscnt 0" ::: "memory");

  // -------- layer 3: (16x64) @ (64x64), 8 WMMAs; keep accumulators --------
  a0 = load_a(st, 0, lane);
  a1 = load_a(st, 1, lane);
  v8f c3[4];
  #pragma unroll
  for (int t = 0; t < 4; ++t) {
    const float bias = b3[t * 16 + nlane];
    v8f acc = {bias, bias, bias, bias, bias, bias, bias, bias};
    acc = __builtin_amdgcn_wmma_f32_16x16x32_f16(false, a0, false, load_b(sW3, 0, t, lane), (short)0, acc, false, false);
    acc = __builtin_amdgcn_wmma_f32_16x16x32_f16(false, a1, false, load_b(sW3, 1, t, lane), (short)0, acc, false, false);
    c3[t] = acc;
  }

  // -------- layer 4: (16x64) @ (64x1) via shuffle reduction --------
  float w4n[4];
  #pragma unroll
  for (int t = 0; t < 4; ++t) w4n[t] = w4[t * 16 + nlane];
  const float b4v = b4[0];

  #pragma unroll
  for (int r = 0; r < 8; ++r) {
    float s = 0.f;
    #pragma unroll
    for (int t = 0; t < 4; ++t) {
      const float v = c3[t][r] > 0.f ? c3[t][r] : 0.f;
      s += v * w4n[t];
    }
    // reduce across the 16-lane N group (xor masks stay within each half of the wave)
    s += __shfl_xor(s, 1);
    s += __shfl_xor(s, 2);
    s += __shfl_xor(s, 4);
    s += __shfl_xor(s, 8);
    if (nlane == 0) {
      const int m = base + r + 8 * hgrp;
      if (m < N) out[m] = s + b4v;
    }
  }
}

extern "C" void kernel_launch(void* const* d_in, const int* in_sizes, int n_in,
                              void* d_out, int out_size, void* d_ws, size_t ws_size,
                              hipStream_t stream) {
  const float* x     = (const float*)d_in[0];
  const float* table = (const float*)d_in[1];
  const float* w1    = (const float*)d_in[2];
  const float* b1    = (const float*)d_in[3];
  const float* w2    = (const float*)d_in[4];
  const float* b2    = (const float*)d_in[5];
  const float* w3    = (const float*)d_in[6];
  const float* b3    = (const float*)d_in[7];
  const float* w4    = (const float*)d_in[8];
  const float* b4    = (const float*)d_in[9];
  float* out = (float*)d_out;

  const int N = in_sizes[0] / 3;
  const int blocks = (N + 127) / 128;  // 8 waves/block * 16 points/wave
  hipLaunchKernelGGL(sdf_fused, dim3(blocks), dim3(256), 0, stream,
                     x, table, w1, b1, w2, b2, w3, b3, w4, b4, out, N);
}
